// LNNEncoder_4552665333880
// MI455X (gfx1250) — compile-verified, matrix operand order checked by reading, not run on hardware
//
#include <hip/hip_runtime.h>
#include <hip/hip_bf16.h>
#include <cstdint>
#include <math.h>

#define B_   32
#define T_   2000
#define FIN_ 80
#define H_   512
#define L_   2

typedef __bf16 bf16;
typedef __attribute__((ext_vector_type(16))) __bf16 v16bf;
typedef __attribute__((ext_vector_type(8)))  __bf16 v8bf;
typedef __attribute__((ext_vector_type(8)))  float  v8f;

// ---------------------------------------------------------------------------
// CDNA5 async copy global->LDS (ASYNCcnt-tracked) helpers
// ---------------------------------------------------------------------------
typedef __attribute__((address_space(3))) const void* lds_cvptr;

__device__ __forceinline__ unsigned lds_addr(const void* p) {
  return (unsigned)(uintptr_t)(lds_cvptr)p;
}

__device__ __forceinline__ void async_copy_b128(const void* lds_dst, const void* gsrc) {
  asm volatile("global_load_async_to_lds_b128 %0, %1, off"
               :: "v"(lds_addr(lds_dst)), "v"(gsrc) : "memory");
}
__device__ __forceinline__ void s_wait_async_le4() {
  asm volatile("s_wait_asynccnt 0x4" ::: "memory");
}
__device__ __forceinline__ void s_wait_async_le0() {
  asm volatile("s_wait_asynccnt 0x0" ::: "memory");
}

// ---------------------------------------------------------------------------
// 0) fp32 -> bf16 transposed weight copies (W_in^T, W_rec^T as [n][k])
// ---------------------------------------------------------------------------
__global__ void cvt_weights_kernel(const float* __restrict__ W_in,
                                   const float* __restrict__ W_rec,
                                   bf16* __restrict__ WinT,
                                   bf16* __restrict__ WrecT) {
  size_t idx = (size_t)blockIdx.x * blockDim.x + threadIdx.x;
  if (idx >= (size_t)L_ * H_ * H_) return;
  int l = (int)(idx / (H_ * H_));
  int r = (int)(idx % (H_ * H_));
  int k = r / H_, n = r % H_;
  size_t o = (size_t)l * H_ * H_ + (size_t)n * H_ + k;
  WinT[o]  = (bf16)W_in[idx];
  WrecT[o] = (bf16)W_rec[idx];
}

// ---------------------------------------------------------------------------
// block-wide sum over 256 threads
// ---------------------------------------------------------------------------
__device__ __forceinline__ float block_sum_256(float v, float* red) {
  int tid = threadIdx.x;
  red[tid] = v;
  __syncthreads();
  #pragma unroll
  for (int s = 128; s > 0; s >>= 1) {
    if (tid < s) red[tid] += red[tid + s];
    __syncthreads();
  }
  float r = red[0];
  __syncthreads();
  return r;
}

// ---------------------------------------------------------------------------
// 1) input projection + LayerNorm + exact GELU -> seq bf16 [T,B,H]
// ---------------------------------------------------------------------------
__global__ void __launch_bounds__(256) proj_ln_gelu_kernel(
    const float* __restrict__ x, const float* __restrict__ Wp,
    const float* __restrict__ bp, const float* __restrict__ g1,
    const float* __restrict__ b1, bf16* __restrict__ seq) {
  __shared__ float xr[FIN_];
  __shared__ float red[256];
  const int tid = threadIdx.x;
  const int row = blockIdx.x;              // row = b*T + t
  const int b = row / T_, t = row % T_;
  const float* xp = x + (size_t)row * FIN_;
  if (tid < FIN_) xr[tid] = xp[tid];
  __syncthreads();
  float acc[2];
  #pragma unroll
  for (int c = 0; c < 2; ++c) {
    int h = tid + 256 * c;
    float s = bp[h];
    #pragma unroll 8
    for (int f = 0; f < FIN_; ++f) s = fmaf(xr[f], Wp[f * H_ + h], s);
    acc[c] = s;
  }
  float mean = block_sum_256(acc[0] + acc[1], red) * (1.0f / H_);
  float d0 = acc[0] - mean, d1 = acc[1] - mean;
  float var = block_sum_256(d0 * d0 + d1 * d1, red) * (1.0f / H_);
  float rstd = rsqrtf(var + 1e-5f);
  bf16* op = seq + ((size_t)t * B_ + b) * H_;
  #pragma unroll
  for (int c = 0; c < 2; ++c) {
    int h = tid + 256 * c;
    float y = (acc[c] - mean) * rstd * g1[h] + b1[h];
    float ge = 0.5f * y * (1.0f + erff(y * 0.70710678118654752f));
    op[h] = (bf16)ge;
  }
}

// ---------------------------------------------------------------------------
// 2) xz GEMM: [T*B,512] (bf16) @ W_in^T-layout (bf16) + bias -> fp32 [T*B,512]
//    WMMA bf16, LDS double-buffered via async global->LDS copies.
// ---------------------------------------------------------------------------
#define MBLK 128
#define NBLK 128
#define KC   32
#define LDSTR 40   // padded LDS row stride in bf16 elements (80 B)

__global__ void __launch_bounds__(256) xz_gemm_kernel(
    const bf16* __restrict__ A,      // [T*B, H]
    const bf16* __restrict__ Bt,     // [H(n), H(k)]
    const float* __restrict__ bias,  // [H]
    float* __restrict__ C) {         // [T*B, H]
  __shared__ __align__(16) bf16 lds[2 * 2 * MBLK * LDSTR];  // 40 KB
  const int tid = threadIdx.x;
  const int lane = tid & 31, wid = tid >> 5;
  const int wm = wid & 3, wn = wid >> 2;   // 4(M) x 2(N) wave grid
  const int lr = lane & 15, kh = lane >> 4;
  const int row0 = blockIdx.y * MBLK;
  const int n0 = blockIdx.x * NBLK;

  auto stage = [&](int kc, int buf) {
    bf16* dstA = lds + buf * (2 * MBLK * LDSTR);
    bf16* dstB = dstA + MBLK * LDSTR;
    const int k0 = kc * KC;
    #pragma unroll
    for (int i = 0; i < 2; ++i) {
      int idx = tid + i * 256;           // 0..511 : 128 rows x 4 16B-chunks
      int r = idx >> 2, c = idx & 3;
      async_copy_b128(dstA + r * LDSTR + c * 8,
                      A + (size_t)(row0 + r) * H_ + k0 + c * 8);
      async_copy_b128(dstB + r * LDSTR + c * 8,
                      Bt + (size_t)(n0 + r) * H_ + k0 + c * 8);
    }
  };

  v8f zero = {};
  v8f acc[2][4];
  #pragma unroll
  for (int f = 0; f < 2; ++f)
    #pragma unroll
    for (int g = 0; g < 4; ++g) acc[f][g] = zero;

  stage(0, 0);
  const int NK = H_ / KC;  // 16
  for (int kc = 0; kc < NK; ++kc) {
    const int buf = kc & 1;
    if (kc + 1 < NK) { stage(kc + 1, buf ^ 1); s_wait_async_le4(); }
    else             { s_wait_async_le0(); }
    __syncthreads();
    const bf16* As = lds + buf * (2 * MBLK * LDSTR);
    const bf16* Bs = As + MBLK * LDSTR;
    union Frag { v16bf v; v8bf h[2]; };
    Frag a[2], bb[4];
    #pragma unroll
    for (int f = 0; f < 2; ++f) {
      const bf16* p = As + (wm * 32 + f * 16 + lr) * LDSTR + 8 * kh;
      a[f].h[0] = *(const v8bf*)p;
      a[f].h[1] = *(const v8bf*)(p + 16);
    }
    #pragma unroll
    for (int g = 0; g < 4; ++g) {
      const bf16* p = Bs + (wn * 64 + g * 16 + lr) * LDSTR + 8 * kh;
      bb[g].h[0] = *(const v8bf*)p;
      bb[g].h[1] = *(const v8bf*)(p + 16);
    }
    #pragma unroll
    for (int f = 0; f < 2; ++f)
      #pragma unroll
      for (int g = 0; g < 4; ++g)
        acc[f][g] = __builtin_amdgcn_wmma_f32_16x16x32_bf16(
            false, a[f].v, false, bb[g].v, (short)0, acc[f][g], false, false);
    __syncthreads();
  }
  #pragma unroll
  for (int f = 0; f < 2; ++f)
    #pragma unroll
    for (int g = 0; g < 4; ++g) {
      const int col = n0 + wn * 64 + g * 16 + lr;
      const float bv = bias[col];
      #pragma unroll
      for (int r = 0; r < 8; ++r) {
        const int row = row0 + wm * 32 + f * 16 + r + 8 * kh;
        C[(size_t)row * H_ + col] = acc[f][g][r] + bv;
      }
    }
}

// ---------------------------------------------------------------------------
// 3) LTC scan: batch-parallel (2 WGs x 16 batch rows), h state in LDS
//    (fp32 master + bf16 WMMA copy), 16 x wmma_bf16 per step per wave.
// ---------------------------------------------------------------------------
__global__ void __launch_bounds__(1024) ltc_scan_kernel(
    const float* __restrict__ xz,       // [T,B,H] fp32 input drive
    float* ys,                          // alias of xz; written iff write_ys
    const bf16* __restrict__ WrT,       // [H(n), H(k)] bf16 (this layer)
    const float* __restrict__ tau_raw,  // [H]
    bf16* seq_out,                      // [T,B,H] bf16; written iff write_seq
    float* __restrict__ h_last,         // [B,H] slice of d_out hidden
    int write_seq, int write_ys) {
  __shared__ float h32[16 * H_];                  // 32 KB fp32 master state
  __shared__ __align__(16) bf16 hbf[16 * 520];    // bf16 copy (padded rows)
  const int tid = threadIdx.x;
  const int lane = tid & 31, wid = tid >> 5;      // 32 waves
  const int gb = blockIdx.x;                      // batch rows 16*gb..+15
  const int lr = lane & 15, kh = lane >> 4;
  const int n = wid * 16 + lr;                    // this lane's output column

  float trw = tau_raw[n];
  float sp = (trw > 20.0f) ? trw : log1pf(expf(trw));
  float itau = 1.0f / (sp + 0.1f);

  for (int i = tid; i < 16 * H_; i += 1024) h32[i] = 0.0f;
  for (int i = tid; i < 16 * 520; i += 1024) hbf[i] = (bf16)0.0f;
  __syncthreads();

  for (int t = 0; t < T_; ++t) {
    const size_t rb = ((size_t)t * B_ + 16 * gb) * H_;
    v8f acc;
    #pragma unroll
    for (int r = 0; r < 8; ++r)
      acc[r] = xz[rb + (size_t)(r + 8 * kh) * H_ + n];
    if (t + 1 < T_)
      __builtin_prefetch(xz + rb + (size_t)B_ * H_ + (size_t)(8 * kh) * H_ + n, 0, 0);

    #pragma unroll
    for (int ks = 0; ks < 16; ++ks) {         // K = 512 = 16 x 32
      union { v16bf v; v8bf h[2]; } a, w;
      const bf16* ap = hbf + lr * 520 + ks * 32 + 8 * kh;
      a.h[0] = *(const v8bf*)ap;
      a.h[1] = *(const v8bf*)(ap + 16);
      const bf16* wp = WrT + (size_t)n * H_ + ks * 32 + 8 * kh;
      w.h[0] = *(const v8bf*)wp;
      w.h[1] = *(const v8bf*)(wp + 16);
      acc = __builtin_amdgcn_wmma_f32_16x16x32_bf16(
          false, a.v, false, w.v, (short)0, acc, false, false);
    }
    __syncthreads();   // all reads of hbf for step t are done
    #pragma unroll
    for (int r = 0; r < 8; ++r) {
      int m = r + 8 * kh;                      // row in 16-row tile
      float hold = h32[m * H_ + n];
      float hn = hold + (tanhf(acc[r]) - hold) * itau;
      h32[m * H_ + n] = hn;
      bf16 hb = (bf16)hn;
      hbf[m * 520 + n] = hb;
      if (write_seq) seq_out[rb + (size_t)m * H_ + n] = hb;
      if (write_ys)  ys[rb + (size_t)m * H_ + n] = hn;  // in-place over xz[t]
    }
    __syncthreads();   // h updated before next step's fragment reads
  }
  #pragma unroll
  for (int r = 0; r < 8; ++r) {
    int m = r + 8 * kh;
    h_last[(size_t)(16 * gb + m) * H_ + n] = h32[m * H_ + n];
  }
}

// ---------------------------------------------------------------------------
// 4) final LayerNorm + transpose [T,B,H] -> encoded [B,T,H]
// ---------------------------------------------------------------------------
__global__ void __launch_bounds__(256) final_ln_kernel(
    const float* __restrict__ ysv, const float* __restrict__ g,
    const float* __restrict__ bta, float* __restrict__ out) {
  __shared__ float red[256];
  const int tid = threadIdx.x;
  const int row = blockIdx.x;              // b*T + t
  const int b = row / T_, t = row % T_;
  const float* ip = ysv + ((size_t)t * B_ + b) * H_;
  float v0 = ip[tid], v1 = ip[tid + 256];
  float mean = block_sum_256(v0 + v1, red) * (1.0f / H_);
  float d0 = v0 - mean, d1 = v1 - mean;
  float var = block_sum_256(d0 * d0 + d1 * d1, red) * (1.0f / H_);
  float rstd = rsqrtf(var + 1e-5f);
  float* op = out + (size_t)row * H_;
  op[tid]       = d0 * rstd * g[tid] + bta[tid];
  op[tid + 256] = d1 * rstd * g[tid + 256] + bta[tid + 256];
}

// ---------------------------------------------------------------------------
extern "C" void kernel_launch(void* const* d_in, const int* in_sizes, int n_in,
                              void* d_out, int out_size, void* d_ws, size_t ws_size,
                              hipStream_t stream) {
  (void)in_sizes; (void)n_in; (void)out_size; (void)ws_size;
  const float* x       = (const float*)d_in[0];
  const float* W_proj  = (const float*)d_in[1];
  const float* b_proj  = (const float*)d_in[2];
  const float* ln1_g   = (const float*)d_in[3];
  const float* ln1_b   = (const float*)d_in[4];
  const float* W_in    = (const float*)d_in[5];
  const float* W_rec   = (const float*)d_in[6];
  const float* b_rnn   = (const float*)d_in[7];
  const float* tau_raw = (const float*)d_in[8];
  const float* lno_g   = (const float*)d_in[9];
  const float* lno_b   = (const float*)d_in[10];

  float* out    = (float*)d_out;
  float* hidden = out + (size_t)B_ * T_ * H_;   // [L,B,H]

  char* ws = (char*)d_ws;
  float* xz   = (float*)ws;  ws += (size_t)T_ * B_ * H_ * sizeof(float);   // 131 MB
  bf16*  seq  = (bf16*)ws;   ws += (size_t)T_ * B_ * H_ * sizeof(bf16);    // 65 MB
  bf16*  WinT = (bf16*)ws;   ws += (size_t)L_ * H_ * H_ * sizeof(bf16);
  bf16*  WrecT= (bf16*)ws;

  {
    int tot = L_ * H_ * H_;
    cvt_weights_kernel<<<(tot + 255) / 256, 256, 0, stream>>>(W_in, W_rec, WinT, WrecT);
  }
  proj_ln_gelu_kernel<<<B_ * T_, 256, 0, stream>>>(x, W_proj, b_proj, ln1_g, ln1_b, seq);

  for (int l = 0; l < L_; ++l) {
    dim3 grid(H_ / NBLK, (T_ * B_) / MBLK);   // 4 x 500
    xz_gemm_kernel<<<grid, 256, 0, stream>>>(seq, WinT + (size_t)l * H_ * H_,
                                             b_rnn + l * H_, xz);
    ltc_scan_kernel<<<B_ / 16, 1024, 0, stream>>>(
        xz, xz, WrecT + (size_t)l * H_ * H_, tau_raw + l * H_, seq,
        hidden + (size_t)l * B_ * H_,
        /*write_seq=*/(l < L_ - 1), /*write_ys=*/(l == L_ - 1));
  }
  final_ln_kernel<<<B_ * T_, 256, 0, stream>>>(xz, lno_g, lno_b, out);
}
